// CLSTokenTransformerFusion_22084721836787
// MI455X (gfx1250) — compile-verified, matrix operand order checked by reading, not run on hardware
//
#include <hip/hip_runtime.h>
#include <math.h>

// ---------------- problem constants ----------------
#define DM    512            // model dim
#define FFD   2048           // ff dim
#define NH    8              // heads
#define DH    64             // head dim
#define NL    2              // layers
#define NTOK  4              // tokens per sequence (CLS + 3 modalities)
#define SEQ_PER_WG 16
#define ROWS  (SEQ_PER_WG * NTOK)   // 64 rows per workgroup
#define NSEQ  (32 * 1024)           // B*T

// LDS strides (in ushort elements), padded so byte stride is a multiple of 16
#define XB_STRIDE 520        // 64 x 520  bf16  (current activations)
#define QB_STRIDE 1544       // 64 x 1544 bf16  (qkv / attn-out / gemm staging)
#define HB_STRIDE 136        // 64 x 136  bf16  (ffn hidden chunk)

typedef __attribute__((ext_vector_type(16))) __bf16 v16bf;
typedef __attribute__((ext_vector_type(8)))  float  v8f;

// ---------------- small helpers ----------------
// Native hardware converts (v_cvt_*_bf16 / shift) instead of integer emulation.
static __device__ inline unsigned short f2bf(float f) {
  __bf16 h = (__bf16)f;
  return __builtin_bit_cast(unsigned short, h);
}
static __device__ inline float bf2f(unsigned short u) {
  return (float)__builtin_bit_cast(__bf16, u);
}

// Load one 16x32 bf16 WMMA operand tile from a row-major [R][stride] bf16
// buffer (works for A from LDS and for B from row-major [N][K] weights).
// Per CDNA5 ISA 7.12.2: lane 0-15 -> row, K 0..7 & 16..23; lanes 16-31 -> K+8.
static __device__ inline v16bf load_frag(const unsigned short* base, int stride,
                                         int r0, int k0, int ln) {
  const int r  = ln & 15;
  const int kb = (ln < 16) ? 0 : 8;
  const unsigned short* p = base + (size_t)(r0 + r) * stride + (k0 + kb);
  union { v16bf v; uint4 q[2]; } f;
  f.q[0] = *(const uint4*)(p);        // K = kb .. kb+7
  f.q[1] = *(const uint4*)(p + 16);   // K = kb+16 .. kb+23
  return f.v;
}

static __device__ inline v8f wmma_bf16(v16bf a, v16bf b, v8f c) {
  // (neg_a, A, neg_b, B, c_mod, C, reuse_a, reuse_b)
  return __builtin_amdgcn_wmma_f32_16x16x32_bf16(false, a, false, b,
                                                 (short)0, c, false, false);
}

static __device__ inline v8f bias_acc(float bv) {
  v8f c;
  #pragma unroll
  for (int r = 0; r < 8; ++r) c[r] = bv;
  return c;
}

// Store a 16x16 f32 C tile (8 VGPRs) as bf16 into [rows][stride] buffer.
// C/D layout: lanes 0-15: M=r, N=lane; lanes 16-31: M=r+8, N=lane-16.
static __device__ inline void store_c(unsigned short* base, int stride,
                                      int m0, int n0, int ln, v8f c) {
  const int n  = ln & 15;
  const int mo = (ln < 16) ? 0 : 8;
  unsigned short* p = base + (size_t)(m0 + mo) * stride + (n0 + n);
  #pragma unroll
  for (int r = 0; r < 8; ++r) p[(size_t)r * stride] = f2bf(c[r]);
}

static __device__ inline float wsum(float v) {
  v += __shfl_xor(v, 16, 32);
  v += __shfl_xor(v,  8, 32);
  v += __shfl_xor(v,  4, 32);
  v += __shfl_xor(v,  2, 32);
  v += __shfl_xor(v,  1, 32);
  return v;
}

// Residual + LayerNorm of one row: x = LN(x + y) with gamma/beta, in-place on xr.
static __device__ inline void ln_row(unsigned short* xr, const unsigned short* yr,
                                     const float* g, const float* b, int ln) {
  float s1 = 0.f, s2 = 0.f;
  #pragma unroll 1
  for (int c = ln; c < DM; c += 32) {
    float z = bf2f(xr[c]) + bf2f(yr[c]);
    s1 += z; s2 += z * z;
  }
  s1 = wsum(s1); s2 = wsum(s2);
  const float mean = s1 * (1.f / DM);
  const float var  = s2 * (1.f / DM) - mean * mean;
  const float rstd = rsqrtf(var + 1e-5f);
  #pragma unroll 1
  for (int c = ln; c < DM; c += 32) {
    float z = bf2f(xr[c]) + bf2f(yr[c]);
    xr[c] = f2bf((z - mean) * rstd * g[c] + b[c]);
  }
}

// ---------------- weight fp32 -> bf16 pre-pass ----------------
__global__ void cvt_bf16_kernel(const float* __restrict__ s,
                                unsigned short* __restrict__ d, int n) {
  int i = blockIdx.x * blockDim.x + threadIdx.x;
  if (i < n) d[i] = f2bf(s[i]);
}

// ---------------- fused transformer kernel ----------------
__global__ __launch_bounds__(512, 1)
void cls_fusion_kernel(const float* __restrict__ ecg,
                       const float* __restrict__ ppg,
                       const float* __restrict__ resp,
                       const float* __restrict__ cls,
                       const float* __restrict__ modemb,
                       const unsigned short* __restrict__ wqkv,  // [L][3D][D] bf16
                       const float* __restrict__ bqkv,           // [L][3D]
                       const unsigned short* __restrict__ wout,  // [L][D][D] bf16
                       const float* __restrict__ bout,           // [L][D]
                       const float* __restrict__ g1, const float* __restrict__ b1,
                       const unsigned short* __restrict__ wff1,  // [L][FF][D] bf16
                       const float* __restrict__ bff1,           // [L][FF]
                       const unsigned short* __restrict__ wff2,  // [L][D][FF] bf16
                       const float* __restrict__ bff2,           // [L][D]
                       const float* __restrict__ g2, const float* __restrict__ b2,
                       const float* __restrict__ fg, const float* __restrict__ fb,
                       float* __restrict__ out) {
  extern __shared__ unsigned short smem[];
  unsigned short* xb = smem;                               // [64][520]
  unsigned short* qb = smem + ROWS * XB_STRIDE;            // [64][1544] q|k|v
  unsigned short* hb = qb + ROWS * QB_STRIDE;              // [64][136]

  const int tid = threadIdx.x;
  const int ln  = tid & 31;        // lane in wave32
  const int w   = tid >> 5;        // wave id, 0..15
  const int wgBase = blockIdx.x * SEQ_PER_WG;   // first sequence of this WG

  // ---- Phase 0: gather inputs + modality embedding -> xb (bf16) ----
  #pragma unroll 1
  for (int idx = tid; idx < ROWS * (DM / 4); idx += 512) {
    const int row = idx >> 7;            // /128 float4s per row
    const int c4  = idx & 127;
    const int tok = row & 3;
    const int seq = row >> 2;
    const size_t gs = (size_t)(wgBase + seq);
    const float4* src;
    if (tok == 0)      src = (const float4*)cls;
    else if (tok == 1) src = (const float4*)(ecg  + gs * DM);
    else if (tok == 2) src = (const float4*)(ppg  + gs * DM);
    else               src = (const float4*)(resp + gs * DM);
    const float4 v  = src[c4];
    const float4 me = ((const float4*)(modemb + tok * DM))[c4];
    unsigned short* d = xb + (size_t)row * XB_STRIDE + c4 * 4;
    d[0] = f2bf(v.x + me.x); d[1] = f2bf(v.y + me.y);
    d[2] = f2bf(v.z + me.z); d[3] = f2bf(v.w + me.w);
  }
  __syncthreads();

  #pragma unroll 1
  for (int l = 0; l < NL; ++l) {
    const unsigned short* WQ = wqkv + (size_t)l * (3 * DM * DM);
    const unsigned short* WO = wout + (size_t)l * (DM * DM);
    const unsigned short* W1 = wff1 + (size_t)l * (FFD * DM);
    const unsigned short* W2 = wff2 + (size_t)l * (DM * FFD);
    const float* BQ = bqkv + l * 3 * DM;
    const float* BO = bout + l * DM;
    const float* B1 = bff1 + l * FFD;
    const float* B2 = bff2 + l * DM;
    const float* G1 = g1 + l * DM; const float* Be1 = b1 + l * DM;
    const float* G2 = g2 + l * DM; const float* Be2 = b2 + l * DM;

    // ---- QKV GEMM: [64x512] @ [512x1536] -> qb ----
    // 96 N-tiles total; wave owns 6, processed as 3 passes of 2 to keep
    // register pressure low (acc pair + A + 2 B frags only).
    #pragma unroll 1
    for (int g = 0; g < 3; ++g) {
      const int n0 = (w * 6 + g * 2) * 16;
      #pragma unroll 1
      for (int mt = 0; mt < 4; ++mt) {
        const int m0 = mt * 16;
        v8f acc0 = bias_acc(BQ[n0 + (ln & 15)]);
        v8f acc1 = bias_acc(BQ[n0 + 16 + (ln & 15)]);
        #pragma unroll 1
        for (int kt = 0; kt < 16; ++kt) {
          const v16bf a  = load_frag(xb, XB_STRIDE, m0, kt * 32, ln);
          const v16bf b0 = load_frag(WQ, DM, n0, kt * 32, ln);
          const v16bf b1f = load_frag(WQ, DM, n0 + 16, kt * 32, ln);
          acc0 = wmma_bf16(a, b0, acc0);
          acc1 = wmma_bf16(a, b1f, acc1);
        }
        store_c(qb, QB_STRIDE, m0, n0, ln, acc0);
        store_c(qb, QB_STRIDE, m0, n0 + 16, ln, acc1);
      }
    }
    __syncthreads();

    // ---- Attention: 16 seq x 8 heads = 128 (s,h) pairs; 4 per wave-half ----
    #pragma unroll 1
    for (int it = 0; it < 4; ++it) {
      const int pair = w * 8 + it * 2 + (ln >> 4);
      const int s = pair >> 3, h = pair & 7;
      const int p = ln & 15, qi = p >> 2, kj = p & 3;
      // scores
      const unsigned short* qr = qb + (size_t)(s * 4 + qi) * QB_STRIDE + h * DH;
      const unsigned short* kr = qb + (size_t)(s * 4 + kj) * QB_STRIDE + DM + h * DH;
      float dot = 0.f;
      #pragma unroll 8
      for (int t = 0; t < DH; ++t) dot += bf2f(qr[t]) * bf2f(kr[t]);
      float sc = dot * 0.125f;                       // 1/sqrt(64)
      float mx = sc;
      mx = fmaxf(mx, __shfl_xor(mx, 1, 32));
      mx = fmaxf(mx, __shfl_xor(mx, 2, 32));
      float e = __expf(sc - mx);
      float sum = e;
      sum += __shfl_xor(sum, 1, 32);
      sum += __shfl_xor(sum, 2, 32);
      const float aat = e / sum;
      // output: lane p -> (i = p>>2, dgroup = p&3), gather a[i][0..3] via shfl
      const int half = (ln >> 4) << 4;
      const int oi = p >> 2, dg = p & 3;
      const float a0 = __shfl(aat, half + oi * 4 + 0, 32);
      const float a1 = __shfl(aat, half + oi * 4 + 1, 32);
      const float a2 = __shfl(aat, half + oi * 4 + 2, 32);
      const float a3 = __shfl(aat, half + oi * 4 + 3, 32);
      const unsigned short* v0 = qb + (size_t)(s * 4 + 0) * QB_STRIDE + 2 * DM + h * DH + dg * 16;
      const unsigned short* v1 = qb + (size_t)(s * 4 + 1) * QB_STRIDE + 2 * DM + h * DH + dg * 16;
      const unsigned short* v2 = qb + (size_t)(s * 4 + 2) * QB_STRIDE + 2 * DM + h * DH + dg * 16;
      const unsigned short* v3 = qb + (size_t)(s * 4 + 3) * QB_STRIDE + 2 * DM + h * DH + dg * 16;
      unsigned short* orow = qb + (size_t)(s * 4 + oi) * QB_STRIDE + h * DH + dg * 16;
      #pragma unroll
      for (int d = 0; d < 16; ++d)
        orow[d] = f2bf(a0 * bf2f(v0[d]) + a1 * bf2f(v1[d]) +
                       a2 * bf2f(v2[d]) + a3 * bf2f(v3[d]));
    }
    __syncthreads();

    // ---- Out projection: [64x512]@[512x512] -> staged at qb cols 512.. ----
    {
      const int n0 = (w * 2) * 16;
      #pragma unroll 1
      for (int mt = 0; mt < 4; ++mt) {
        const int m0 = mt * 16;
        v8f acc0 = bias_acc(BO[n0 + (ln & 15)]);
        v8f acc1 = bias_acc(BO[n0 + 16 + (ln & 15)]);
        #pragma unroll 1
        for (int kt = 0; kt < 16; ++kt) {
          const v16bf a  = load_frag(qb, QB_STRIDE, m0, kt * 32, ln);  // attn out
          const v16bf b0 = load_frag(WO, DM, n0, kt * 32, ln);
          const v16bf b1f = load_frag(WO, DM, n0 + 16, kt * 32, ln);
          acc0 = wmma_bf16(a, b0, acc0);
          acc1 = wmma_bf16(a, b1f, acc1);
        }
        store_c(qb + DM, QB_STRIDE, m0, n0, ln, acc0);
        store_c(qb + DM, QB_STRIDE, m0, n0 + 16, ln, acc1);
      }
    }
    __syncthreads();

    // ---- Residual + LN1 (wave w handles rows 4w..4w+3) ----
    #pragma unroll 1
    for (int rr = 0; rr < 4; ++rr) {
      const int row = w * 4 + rr;
      ln_row(xb + (size_t)row * XB_STRIDE,
             qb + (size_t)row * QB_STRIDE + DM, G1, Be1, ln);
    }
    __syncthreads();

    // ---- Fused FFN: chunks of 128 over FF dim; acc2 lives in VGPRs ----
    v8f acc2[4][2];
    #pragma unroll
    for (int mt = 0; mt < 4; ++mt)
      #pragma unroll
      for (int j = 0; j < 2; ++j)
        acc2[mt][j] = bias_acc(B2[(w * 2 + j) * 16 + (ln & 15)]);

    #pragma unroll 1
    for (int f = 0; f < FFD / 128; ++f) {
      // FF1 chunk: h[64x128] = gelu(x @ W1^T[:, f*128 .. ]); 2 tiles per wave,
      // one accumulator live at a time.
      #pragma unroll 1
      for (int j = 0; j < 2; ++j) {
        const int tile = w * 2 + j;
        const int m0 = (tile >> 3) * 16, n0 = (tile & 7) * 16;
        v8f acc = bias_acc(B1[f * 128 + n0 + (ln & 15)]);
        #pragma unroll 1
        for (int kt = 0; kt < 16; ++kt) {
          const v16bf a  = load_frag(xb, XB_STRIDE, m0, kt * 32, ln);
          const v16bf bf = load_frag(W1, DM, f * 128 + n0, kt * 32, ln);
          acc = wmma_bf16(a, bf, acc);
        }
        // exact gelu, store bf16 tile to hb
        const int n  = ln & 15;
        const int mo = (ln < 16) ? 0 : 8;
        unsigned short* p = hb + (size_t)(m0 + mo) * HB_STRIDE + n0 + n;
        #pragma unroll
        for (int r = 0; r < 8; ++r) {
          const float x = acc[r];
          p[(size_t)r * HB_STRIDE] =
              f2bf(0.5f * x * (1.f + erff(x * 0.70710678118f)));
        }
      }
      __syncthreads();
      // FF2 accumulate: acc2 += h @ W2^T chunk
      #pragma unroll
      for (int mt = 0; mt < 4; ++mt) {
        #pragma unroll 1
        for (int kt = 0; kt < 4; ++kt) {
          const v16bf a  = load_frag(hb, HB_STRIDE, mt * 16, kt * 32, ln);
          const v16bf b0 =
              load_frag(W2, FFD, (w * 2 + 0) * 16, f * 128 + kt * 32, ln);
          acc2[mt][0] = wmma_bf16(a, b0, acc2[mt][0]);
          const v16bf b1f =
              load_frag(W2, FFD, (w * 2 + 1) * 16, f * 128 + kt * 32, ln);
          acc2[mt][1] = wmma_bf16(a, b1f, acc2[mt][1]);
        }
      }
      __syncthreads();
    }
    // stage FF2 result, then residual + LN2
    #pragma unroll
    for (int mt = 0; mt < 4; ++mt)
      #pragma unroll
      for (int j = 0; j < 2; ++j)
        store_c(qb + DM, QB_STRIDE, mt * 16, (w * 2 + j) * 16, ln, acc2[mt][j]);
    __syncthreads();
    #pragma unroll 1
    for (int rr = 0; rr < 4; ++rr) {
      const int row = w * 4 + rr;
      ln_row(xb + (size_t)row * XB_STRIDE,
             qb + (size_t)row * QB_STRIDE + DM, G2, Be2, ln);
    }
    __syncthreads();
  }

  // ---- Final: LN of CLS rows (token 0), fp32 output; wave w -> seq w ----
  {
    const int s = w;
    const int row = s * 4;
    const size_t gs = (size_t)(wgBase + s);
    const unsigned short* xr = xb + (size_t)row * XB_STRIDE;
    float s1 = 0.f, s2 = 0.f;
    #pragma unroll 1
    for (int c = ln; c < DM; c += 32) {
      const float z = bf2f(xr[c]);
      s1 += z; s2 += z * z;
    }
    s1 = wsum(s1); s2 = wsum(s2);
    const float mean = s1 * (1.f / DM);
    const float var  = s2 * (1.f / DM) - mean * mean;
    const float rstd = rsqrtf(var + 1e-5f);
    #pragma unroll 1
    for (int c = ln; c < DM; c += 32) {
      const float z = bf2f(xr[c]);
      out[gs * DM + c] = (z - mean) * rstd * fg[c] + fb[c];
    }
  }
}

// ---------------- host launch ----------------
extern "C" void kernel_launch(void* const* d_in, const int* in_sizes, int n_in,
                              void* d_out, int out_size, void* d_ws, size_t ws_size,
                              hipStream_t stream) {
  const float* ecg    = (const float*)d_in[0];
  const float* ppg    = (const float*)d_in[1];
  const float* resp   = (const float*)d_in[2];
  const float* cls    = (const float*)d_in[3];
  const float* modemb = (const float*)d_in[4];
  const float* ipw    = (const float*)d_in[5];
  const float* ipb    = (const float*)d_in[6];
  const float* ow     = (const float*)d_in[7];
  const float* ob     = (const float*)d_in[8];
  const float* ln1g   = (const float*)d_in[9];
  const float* ln1b   = (const float*)d_in[10];
  const float* f1w    = (const float*)d_in[11];
  const float* f1b    = (const float*)d_in[12];
  const float* f2w    = (const float*)d_in[13];
  const float* f2b    = (const float*)d_in[14];
  const float* ln2g   = (const float*)d_in[15];
  const float* ln2b   = (const float*)d_in[16];
  const float* flng   = (const float*)d_in[17];
  const float* flnb   = (const float*)d_in[18];
  float* out = (float*)d_out;

  // bf16 weight layout in workspace
  const int nqkv = NL * 3 * DM * DM;   // 1,572,864
  const int nout = NL * DM * DM;       //   524,288
  const int nff1 = NL * FFD * DM;      // 2,097,152
  const int nff2 = NL * DM * FFD;      // 2,097,152
  unsigned short* ws   = (unsigned short*)d_ws;
  unsigned short* wqkv = ws;
  unsigned short* wout = wqkv + nqkv;
  unsigned short* wff1 = wout + nout;
  unsigned short* wff2 = wff1 + nff1;

  cvt_bf16_kernel<<<(nqkv + 255) / 256, 256, 0, stream>>>(ipw, wqkv, nqkv);
  cvt_bf16_kernel<<<(nout + 255) / 256, 256, 0, stream>>>(ow,  wout, nout);
  cvt_bf16_kernel<<<(nff1 + 255) / 256, 256, 0, stream>>>(f1w, wff1, nff1);
  cvt_bf16_kernel<<<(nff2 + 255) / 256, 256, 0, stream>>>(f2w, wff2, nff2);

  const size_t smem =
      (size_t)(ROWS * XB_STRIDE + ROWS * QB_STRIDE + ROWS * HB_STRIDE) *
      sizeof(unsigned short);   // ~275 KB, under 320 KB/WGP

  cls_fusion_kernel<<<NSEQ / SEQ_PER_WG, 512, smem, stream>>>(
      ecg, ppg, resp, cls, modemb,
      wqkv, ipb, wout, ob, ln1g, ln1b,
      wff1, f1b, wff2, f2b, ln2g, ln2b,
      flng, flnb, out);
}
